// GAT_7327214207516
// MI455X (gfx1250) — matrix operand order repaired and checked by reading, not compiled, not run on hardware
//
#include <hip/hip_runtime.h>
#include <stdint.h>

typedef __attribute__((ext_vector_type(16))) __bf16 v16bf;
typedef __attribute__((ext_vector_type(8)))  float  v8f;

#define HEADS 8
#define HID   32
#define DIM   256

__device__ __forceinline__ unsigned short f2bf(float f) {
  unsigned u = __float_as_uint(f);
  u += 0x7FFFu + ((u >> 16) & 1u);        // round-to-nearest-even truncation
  return (unsigned short)(u >> 16);
}

__device__ __forceinline__ v16bf ld2x16B(const unsigned short* lo, const unsigned short* hi) {
  union { uint4 q[2]; v16bf v; } u;
  u.q[0] = *reinterpret_cast<const uint4*>(lo);
  u.q[1] = *reinterpret_cast<const uint4*>(hi);
  return u.v;
}

// ---------------- conversion / packing ----------------

__global__ void k_cvt_bf16(const float* __restrict__ src, unsigned short* __restrict__ dst, int n) {
  int t = blockIdx.x * blockDim.x + threadIdx.x;
  if (t < n) dst[t] = f2bf(src[t]);
}

// Pack W [256 x 256] (K x N, row-major f32) into WMMA B-operand tile order:
// Wp[((kt*16+nt)*32+lane)*16 + i] = bf16( W[(kt*32 + (lane<16?0:16) + i)*256 + nt*16 + lane%16] )
__global__ void k_pack_w(const float* __restrict__ W, unsigned short* __restrict__ Wp) {
  int t = blockIdx.x * blockDim.x + threadIdx.x;   // 8*16*32 = 4096 threads
  if (t >= 8 * 16 * 32) return;
  int kt = t >> 9, rem = t & 511, nt = rem >> 5, lane = rem & 31;
  int ncol  = nt * 16 + (lane & 15);
  int kbase = kt * 32 + ((lane < 16) ? 0 : 16);
  unsigned short* out = Wp + (size_t)t * 16;
#pragma unroll
  for (int i = 0; i < 16; ++i) out[i] = f2bf(W[(size_t)(kbase + i) * DIM + ncol]);
}

// ---------------- GEMM: C[M,256] = A[M,256] @ W[256,256], bf16 WMMA, f32 accum ----
// One wave computes a 16x64 strip (4 accumulator tiles). block = 4 waves.
__global__ __launch_bounds__(128) void k_gemm_bf16(const unsigned short* __restrict__ A,
                                                   const unsigned short* __restrict__ Bp,
                                                   float* __restrict__ C) {
  const int lane = threadIdx.x & 31;
  const int wave = threadIdx.x >> 5;       // 0..3 -> ntile group
  const int m0   = blockIdx.x * 16;
  const int nt0  = wave * 4;
  const int m    = m0 + (lane & 15);
  const int kb   = (lane < 16) ? 0 : 16;   // A-layout half-wave K base

  v8f acc[4] = {};
#pragma unroll
  for (int kt = 0; kt < 8; ++kt) {
    const unsigned short* pa = A + (size_t)m * DIM + kt * 32 + kb;
    v16bf a = ld2x16B(pa, pa + 8);         // K={kb..kb+15} contiguous
#pragma unroll
    for (int j = 0; j < 4; ++j) {
      const unsigned short* pb = Bp + (size_t)(((kt * 16 + nt0 + j) * 32) + lane) * 16;
      v16bf b = ld2x16B(pb, pb + 8);       // contiguous 32B packed operand
      acc[j] = __builtin_amdgcn_wmma_f32_16x16x32_bf16(
          false, a, false, b, (short)0, acc[j], false, false);
    }
  }
  const int roff = (lane < 16) ? 0 : 8;
  const int col  = lane & 15;
#pragma unroll
  for (int j = 0; j < 4; ++j)
#pragma unroll
    for (int r = 0; r < 8; ++r)
      C[(size_t)(m0 + roff + r) * DIM + (nt0 + j) * 16 + col] = acc[j][r];
}

// ---------------- attention coefficients ----------------

__global__ void k_attn(const float* __restrict__ h, const float* __restrict__ asv,
                       const float* __restrict__ adv, float* __restrict__ als,
                       float* __restrict__ ald, int nnodes) {
  int t = blockIdx.x * blockDim.x + threadIdx.x;
  if (t >= nnodes * HEADS) return;
  int node = t >> 3, hd = t & 7;
  const float* hp = h + (size_t)node * DIM + hd * HID;
  const float* av = asv + hd * HID;
  const float* bv = adv + hd * HID;
  float s = 0.f, d = 0.f;
#pragma unroll
  for (int c = 0; c < HID; ++c) { float v = hp[c]; s += v * av[c]; d += v * bv[c]; }
  als[t] = s; ald[t] = d;
}

// ---------------- edge phase ----------------

__device__ __forceinline__ void edge_sd(int e, const int* __restrict__ ei, int E, int* s, int* d) {
  if (e < E) { *s = ei[e]; *d = ei[E + e]; } else { *s = *d = e - E; } // self-loops appended
}
__device__ __forceinline__ unsigned fkey(float f) {          // order-preserving f32->u32
  unsigned u = __float_as_uint(f);
  return (u & 0x80000000u) ? ~u : (u | 0x80000000u);
}
__device__ __forceinline__ float funkey(unsigned k) {
  unsigned u = (k & 0x80000000u) ? (k ^ 0x80000000u) : ~k;
  return __uint_as_float(u);
}

__global__ void k_fill_u32(unsigned* __restrict__ p, unsigned v, int n) {
  int t = blockIdx.x * blockDim.x + threadIdx.x;
  if (t < n) p[t] = v;
}

__global__ void k_edge_max(const int* __restrict__ ei, int E, int EP,
                           const float* __restrict__ als, const float* __restrict__ ald,
                           unsigned* __restrict__ mkey) {
  int t = blockIdx.x * blockDim.x + threadIdx.x;
  if (t >= EP * HEADS) return;
  int e = t >> 3, hd = t & 7, s, d;
  edge_sd(e, ei, E, &s, &d);
  float x = als[s * HEADS + hd] + ald[d * HEADS + hd];
  x = (x > 0.f) ? x : 0.2f * x;            // leaky_relu(0.2)
  atomicMax(&mkey[d * HEADS + hd], fkey(x));
}

__global__ void k_edge_exp(const int* __restrict__ ei, int E, int EP,
                           const float* __restrict__ als, const float* __restrict__ ald,
                           const unsigned* __restrict__ mkey, float* __restrict__ z,
                           float* __restrict__ ee) {
  int t = blockIdx.x * blockDim.x + threadIdx.x;
  if (t >= EP * HEADS) return;
  int e = t >> 3, hd = t & 7, s, d;
  edge_sd(e, ei, E, &s, &d);
  float x = als[s * HEADS + hd] + ald[d * HEADS + hd];
  x = (x > 0.f) ? x : 0.2f * x;
  float m = funkey(mkey[d * HEADS + hd]);
  float v = __expf(x - m);
  ee[t] = v;
  atomicAdd(&z[d * HEADS + hd], v);
}

// Normalize in place: ee[e,h] <- ee[e,h] / z[dst(e),h]   (hoists div + z-load out of scatter)
__global__ void k_edge_norm(const int* __restrict__ ei, int E, int EP,
                            const float* __restrict__ z, float* __restrict__ ee) {
  int t = blockIdx.x * blockDim.x + threadIdx.x;
  if (t >= EP * HEADS) return;
  int e = t >> 3, hd = t & 7, s, d;
  edge_sd(e, ei, E, &s, &d);
  ee[t] = ee[t] / z[d * HEADS + hd];
}

// Scatter: agg[dst,h,c] += alpha[e,h] * h[src,h,c].  One thread per (edge, 4 channels):
// vectorized b128 source load, 4 f32 atomics. 64 threads/edge.
__global__ void k_edge_aggr(const int* __restrict__ ei, int E, int EP,
                            const float* __restrict__ ee,
                            const float* __restrict__ h, float* __restrict__ agg) {
  int t = blockIdx.x * blockDim.x + threadIdx.x;   // EP*64 = 54.4M
  if (t >= EP * 64) return;
  int e = t >> 6, q = t & 63, hd = q >> 3, s, d;
  edge_sd(e, ei, E, &s, &d);
  float w = ee[e * HEADS + hd];
  const float4 hv = *reinterpret_cast<const float4*>(h + (size_t)s * DIM + q * 4);
  float* dst = agg + (size_t)d * DIM + q * 4;
  atomicAdd(dst + 0, w * hv.x);
  atomicAdd(dst + 1, w * hv.y);
  atomicAdd(dst + 2, w * hv.z);
  atomicAdd(dst + 3, w * hv.w);
}

// ---------------- finalization ----------------

__global__ void k_relu_bias_cvt(const float* __restrict__ agg, const float* __restrict__ b,
                                unsigned short* __restrict__ hb, int n) {
  int t = blockIdx.x * blockDim.x + threadIdx.x;
  if (t >= n) return;
  float v = agg[t] + b[t & 255];
  v = (v > 0.f) ? v : 0.f;
  hb[t] = f2bf(v);
}

__global__ void k_mean_bias(const float* __restrict__ agg, const float* __restrict__ b2,
                            float* __restrict__ out, int nnodes) {
  int t = blockIdx.x * blockDim.x + threadIdx.x;
  if (t >= nnodes * HID) return;
  int node = t >> 5, c = t & 31;
  float s = 0.f;
#pragma unroll
  for (int hd = 0; hd < HEADS; ++hd) s += agg[(size_t)node * DIM + hd * HID + c];
  out[t] = s * (1.0f / HEADS) + b2[c];
}

// ---------------- launcher ----------------

extern "C" void kernel_launch(void* const* d_in, const int* in_sizes, int n_in,
                              void* d_out, int out_size, void* d_ws, size_t ws_size,
                              hipStream_t stream) {
  const float* x   = (const float*)d_in[0];
  const int*   ei  = (const int*)  d_in[1];
  const float* W1  = (const float*)d_in[2];
  const float* as1 = (const float*)d_in[3];
  const float* ad1 = (const float*)d_in[4];
  const float* b1  = (const float*)d_in[5];
  const float* W2  = (const float*)d_in[6];
  const float* as2 = (const float*)d_in[7];
  const float* ad2 = (const float*)d_in[8];
  const float* b2  = (const float*)d_in[9];

  const int Nn = in_sizes[0] / DIM;   // 50000 (divisible by 16)
  const int E  = in_sizes[1] / 2;     // 800000
  const int EP = E + Nn;              // + self-loops

  size_t off = 0;
  auto carve = [&](size_t bytes) -> void* {
    void* p = (char*)d_ws + off;
    off += (bytes + 255) & ~(size_t)255;
    return p;
  };
  unsigned short* xb   = (unsigned short*)carve((size_t)Nn * DIM * 2); // x / relu(h1) in bf16
  unsigned short* Wp1  = (unsigned short*)carve((size_t)DIM * DIM * 2);
  unsigned short* Wp2  = (unsigned short*)carve((size_t)DIM * DIM * 2);
  float*    hpre = (float*)   carve((size_t)Nn * DIM * 4);
  float*    agg  = (float*)   carve((size_t)Nn * DIM * 4);
  float*    als  = (float*)   carve((size_t)Nn * HEADS * 4);
  float*    ald  = (float*)   carve((size_t)Nn * HEADS * 4);
  unsigned* mkey = (unsigned*)carve((size_t)Nn * HEADS * 4);
  float*    zz   = (float*)   carve((size_t)Nn * HEADS * 4);
  float*    ee   = (float*)   carve((size_t)EP * HEADS * 4);

  const int BT = 256;
  auto cdiv = [](long long a, long long b) { return (int)((a + b - 1) / b); };

  const int nd = Nn * DIM;     // 12.8M
  const int nh = Nn * HEADS;   // 400K
  const int eh = EP * HEADS;   // 6.8M
  const int e4 = EP * 64;      // 54.4M

  k_cvt_bf16<<<cdiv(nd, BT), BT, 0, stream>>>(x, xb, nd);
  k_pack_w <<<cdiv(4096, BT), BT, 0, stream>>>(W1, Wp1);
  k_pack_w <<<cdiv(4096, BT), BT, 0, stream>>>(W2, Wp2);

  for (int layer = 0; layer < 2; ++layer) {
    const unsigned short* Wp = layer ? Wp2 : Wp1;
    const float* asv = layer ? as2 : as1;
    const float* adv = layer ? ad2 : ad1;

    k_gemm_bf16<<<Nn / 16, 128, 0, stream>>>(xb, Wp, hpre);
    k_attn     <<<cdiv(nh, BT), BT, 0, stream>>>(hpre, asv, adv, als, ald, Nn);

    k_fill_u32 <<<cdiv(nh, BT), BT, 0, stream>>>(mkey, 0u, nh);
    k_fill_u32 <<<cdiv(nh, BT), BT, 0, stream>>>((unsigned*)zz, 0u, nh);
    k_fill_u32 <<<cdiv(nd, BT), BT, 0, stream>>>((unsigned*)agg, 0u, nd);

    k_edge_max <<<cdiv(eh, BT), BT, 0, stream>>>(ei, E, EP, als, ald, mkey);
    k_edge_exp <<<cdiv(eh, BT), BT, 0, stream>>>(ei, E, EP, als, ald, mkey, zz, ee);
    k_edge_norm<<<cdiv(eh, BT), BT, 0, stream>>>(ei, E, EP, zz, ee);
    k_edge_aggr<<<cdiv(e4, BT), BT, 0, stream>>>(ei, E, EP, ee, hpre, agg);

    if (layer == 0) {
      k_relu_bias_cvt<<<cdiv(nd, BT), BT, 0, stream>>>(agg, b1, xb, nd);
    } else {
      k_mean_bias<<<cdiv(Nn * HID, BT), BT, 0, stream>>>(agg, b2, (float*)d_out, Nn);
    }
  }
  (void)n_in; (void)out_size; (void)ws_size;
}
